// FastRCNN_146028888279
// MI455X (gfx1250) — matrix operand order repaired and checked by reading, not compiled, not run on hardware
//
#include <hip/hip_runtime.h>
#include <stdint.h>

// ---------------- problem constants ----------------
#define ROI_OS     7
#define ROI_SCALE  0.0625f
#define FEAT_H     50
#define FEAT_W     50
#define FEAT_C     256
#define D_IN       12544          // 256 * 7 * 7
#define N_ROIS     1024           // 2 * 512
#define N_BOX      324            // 4 * 81
#define N_CLS      81

// CDNA5 async global->LDS path (guarded; sync fallback keeps the build green)
#if defined(__HIP_DEVICE_COMPILE__) && defined(__has_builtin)
#if __has_builtin(__builtin_amdgcn_global_load_async_to_lds_b128) && \
    __has_builtin(__builtin_amdgcn_s_wait_asynccnt)
#define USE_ASYNC_LDS 1
#endif
#endif
#ifndef USE_ASYNC_LDS
#define USE_ASYNC_LDS 0
#endif

typedef __attribute__((ext_vector_type(16))) __bf16 v16bf;
typedef __attribute__((ext_vector_type(8)))  float  v8f;
typedef __attribute__((ext_vector_type(4)))  int    vec4i;

#define AS_GLOBAL __attribute__((address_space(1)))
#define AS_LDS    __attribute__((address_space(3)))

static __device__ __forceinline__ __bf16 f2bf(float x) {
  union { float f; uint32_t u; } v; v.f = x;
  uint32_t r = (v.u + 0x7FFFu + ((v.u >> 16) & 1u)) >> 16;   // RNE
  union { unsigned short s; __bf16 b; } o; o.s = (unsigned short)r;
  return o.b;
}

// ---------------- RoI max-pool (torchvision-quantized), f32 -> bf16 ----------------
__global__ __launch_bounds__(256)
void roi_pool_kernel(const float* __restrict__ feats,   // [2,256,50,50]
                     const float* __restrict__ props,   // [2,512,4] flattened
                     __bf16* __restrict__ Xbf)          // [1024, 12544]
{
  const int n = blockIdx.x;          // roi index
  const int c = threadIdx.x;         // channel
  const int img = n >> 9;            // n / 512

  const float* roi = props + (size_t)n * 4;
  const int x1 = (int)rintf(roi[0] * ROI_SCALE);
  const int y1 = (int)rintf(roi[1] * ROI_SCALE);
  const int x2 = (int)rintf(roi[2] * ROI_SCALE);
  const int y2 = (int)rintf(roi[3] * ROI_SCALE);
  const float roi_w = fmaxf((float)(x2 - x1 + 1), 1.0f);
  const float roi_h = fmaxf((float)(y2 - y1 + 1), 1.0f);
  const float bin_w = roi_w / (float)ROI_OS;
  const float bin_h = roi_h / (float)ROI_OS;

  int hs[ROI_OS], he[ROI_OS], wst[ROI_OS], wen[ROI_OS];
#pragma unroll
  for (int p = 0; p < ROI_OS; ++p) {
    int a = (int)floorf((float)p * bin_h) + y1;
    int b = (int)ceilf((float)(p + 1) * bin_h) + y1;
    hs[p] = min(max(a, 0), FEAT_H);
    he[p] = min(max(b, 0), FEAT_H);
    a = (int)floorf((float)p * bin_w) + x1;
    b = (int)ceilf((float)(p + 1) * bin_w) + x1;
    wst[p] = min(max(a, 0), FEAT_W);
    wen[p] = min(max(b, 0), FEAT_W);
  }

  const float* f = feats + ((size_t)img * FEAT_C + c) * (FEAT_H * FEAT_W);
  __bf16* out = Xbf + (size_t)n * D_IN + (size_t)c * (ROI_OS * ROI_OS);

  for (int oh = 0; oh < ROI_OS; ++oh) {
    for (int ow = 0; ow < ROI_OS; ++ow) {
      const int h0 = hs[oh], h1 = he[oh], w0 = wst[ow], w1 = wen[ow];
      float m = -3.402823466e+38f;
      for (int h = h0; h < h1; ++h) {
        const float* row = f + h * FEAT_W;
        for (int w = w0; w < w1; ++w) m = fmaxf(m, row[w]);
      }
      const bool empty = (h1 <= h0) || (w1 <= w0);
      out[oh * ROI_OS + ow] = f2bf(empty ? 0.0f : m);
    }
  }
}

// ---------------- f32 -> bf16 weight convert with column zero-padding ----------------
__global__ __launch_bounds__(256)
void conv_pad_kernel(const float* __restrict__ src, __bf16* __restrict__ dst,
                     int K, int N, int Np)
{
  const int i = blockIdx.x * 256 + threadIdx.x;
  if (i >= K * Np) return;
  const int k = i / Np, n = i - k * Np;
  dst[i] = (n < N) ? f2bf(src[(size_t)k * N + n]) : f2bf(0.0f);
}

// ---------------- tile staging helpers ----------------
// A tile: 128 rows x 32 bf16 = 512 x 16B segments; 2 per thread.
// Async path: global -> LDS directly (ASYNCcnt), no VGPR round-trip.
static __device__ __forceinline__ void stage_a(const __bf16* __restrict__ A, __bf16* asbuf,
                                               int mBase, int K, int kt, int tid)
{
#pragma unroll
  for (int s2 = 0; s2 < 2; ++s2) {
    const int s = tid + s2 * 256;
    const int row = s >> 2, seg = s & 3;
    const __bf16* g = A + (size_t)(mBase + row) * K + kt + seg * 8;
    __bf16* l = asbuf + row * 32 + seg * 8;
#if USE_ASYNC_LDS
    __builtin_amdgcn_global_load_async_to_lds_b128(
        (AS_GLOBAL vec4i*)g, (AS_LDS vec4i*)l, 0, 0);
#else
    *(uint4*)l = *(const uint4*)g;
#endif
  }
}

// B tile: 32 rows x 64 bf16, stored TRANSPOSED ([n][k]) -> needs VGPR round trip.
static __device__ __forceinline__ void stage_b(const __bf16* __restrict__ B, __bf16* bbuf,
                                               int nBase, int Np, int kt, int tid)
{
  const int krow = tid >> 3, seg = tid & 7;          // n0 = seg*8
  union { uint4 q; __bf16 h[8]; } u;
  u.q = *(const uint4*)(B + (size_t)(kt + krow) * Np + nBase + seg * 8);
#pragma unroll
  for (int e = 0; e < 8; ++e)
    bbuf[(seg * 8 + e) * 32 + krow] = u.h[e];
}

// ---------------- bf16 WMMA GEMM: C[1024, Nout] = A[1024,K] * B[K,Np] (+bias, relu) ----
// block = 256 threads (8 waves); tile 128x64, BK = 32; double-buffered LDS.
// wave grid 4(M) x 2(N); each wave: 32x32 subtile = 2x2 wmma accumulators.
template<bool RELU, bool OUT_BF16>
__global__ __launch_bounds__(256)
void gemm_bf16_wmma(const __bf16* __restrict__ A, const __bf16* __restrict__ B,
                    const float* __restrict__ bias, void* __restrict__ Cout,
                    int K, int Np, int Nout)
{
  __shared__ __bf16 As[2][128 * 32];    // [buf][row][k]
  __shared__ __bf16 Bst[2][64 * 32];    // [buf][n][k] (transposed)

  const int tid  = threadIdx.x;
  const int lane = tid & 31;
  const int wid  = tid >> 5;
  const int wm   = wid & 3;             // 0..3 -> M strip of 32
  const int wn   = wid >> 2;            // 0..1 -> N strip of 32
  const int l    = lane & 15;
  const int hi   = lane >> 4;
  const int mBase = blockIdx.y * 128;
  const int nBase = blockIdx.x * 64;

  const v8f vzero = {0.f, 0.f, 0.f, 0.f, 0.f, 0.f, 0.f, 0.f};
  v8f acc[2][2];
#pragma unroll
  for (int m = 0; m < 2; ++m)
#pragma unroll
    for (int n = 0; n < 2; ++n) acc[m][n] = vzero;

  // prologue: stage buffer 0
  stage_a(A, As[0], mBase, K, 0, tid);
  stage_b(B, Bst[0], nBase, Np, 0, tid);

  for (int kt = 0; kt < K; kt += 32) {
    const int  cur  = (kt >> 5) & 1;
    const bool more = (kt + 32) < K;
    if (more) {                          // prefetch next tile into other buffer
      stage_a(A, As[cur ^ 1], mBase, K, kt + 32, tid);
      stage_b(B, Bst[cur ^ 1], nBase, Np, kt + 32, tid);
    }
#if USE_ASYNC_LDS
    // current buffer's 2 async ops complete; next buffer's 2 may stay in flight
    if (more) __builtin_amdgcn_s_wait_asynccnt(2);
    else      __builtin_amdgcn_s_wait_asynccnt(0);
#endif
    __syncthreads();                     // covers B ds_stores (DScnt) + cross-wave visibility

    // ---- fragments per ISA 16-bit layouts, 4 WMMAs ----
    union Frag { v16bf v; uint32_t u[8]; };
    Frag a[2], b[2];
#pragma unroll
    for (int m = 0; m < 2; ++m) {
      const uint32_t* ap = (const uint32_t*)(&As[cur][(wm * 32 + m * 16 + l) * 32]);
#pragma unroll
      for (int v = 0; v < 8; ++v) {
        const int group = v >> 2, sub = v & 3;      // K = group*16 + hi*8 + sub*2
        a[m].u[v] = ap[group * 8 + hi * 4 + sub];
      }
    }
#pragma unroll
    for (int n = 0; n < 2; ++n) {
      const uint32_t* bp = (const uint32_t*)(&Bst[cur][(wn * 32 + n * 16 + l) * 32]);
#pragma unroll
      for (int v = 0; v < 8; ++v)                   // K = (v>>2)*16 + hi*8 + (v&3)*2
        b[n].u[v] = bp[(v >> 2) * 8 + hi * 4 + (v & 3)];
    }
#pragma unroll
    for (int m = 0; m < 2; ++m)
#pragma unroll
      for (int n = 0; n < 2; ++n)
        acc[m][n] = __builtin_amdgcn_wmma_f32_16x16x32_bf16(
            false, a[m].v, false, b[n].v, (short)0, acc[m][n], false, false);

    __syncthreads();                     // all waves done reading `cur` before it is re-staged
  }

  // ---- epilogue: bias (+relu), store (f32 C/D layout: VGPR r -> M = hi*8+r, N = l) ----
#pragma unroll
  for (int m = 0; m < 2; ++m) {
#pragma unroll
    for (int n = 0; n < 2; ++n) {
      const int col = nBase + wn * 32 + n * 16 + l;
      if (col < Nout) {
        const float bv = bias[col];
#pragma unroll
        for (int r = 0; r < 8; ++r) {
          const int row = mBase + wm * 32 + m * 16 + hi * 8 + r;
          float v = acc[m][n][r] + bv;
          if (RELU) v = fmaxf(v, 0.0f);
          if (OUT_BF16)
            ((__bf16*)Cout)[(size_t)row * Nout + col] = f2bf(v);
          else
            ((float*)Cout)[(size_t)row * Nout + col] = v;
        }
      }
    }
  }
}

// ---------------- launch ----------------
extern "C" void kernel_launch(void* const* d_in, const int* in_sizes, int n_in,
                              void* d_out, int out_size, void* d_ws, size_t ws_size,
                              hipStream_t stream) {
  const float* feats = (const float*)d_in[0];
  const float* props = (const float*)d_in[1];
  const float* W1    = (const float*)d_in[2];
  const float* b1    = (const float*)d_in[3];
  const float* W2    = (const float*)d_in[4];
  const float* b2    = (const float*)d_in[5];
  const float* Wbox  = (const float*)d_in[6];
  const float* bbox  = (const float*)d_in[7];
  const float* Wcls  = (const float*)d_in[8];
  const float* bcls  = (const float*)d_in[9];

  // workspace layout (bytes)
  char* ws = (char*)d_ws;
  __bf16* Xbf    = (__bf16*)(ws + 0);           // 1024 x 12544 bf16
  __bf16* W1bf   = (__bf16*)(ws + 25690112);    // 12544 x 1024
  __bf16* H1bf   = (__bf16*)(ws + 51380224);    // 1024 x 1024
  __bf16* W2bf   = (__bf16*)(ws + 53477376);    // 1024 x 1024
  __bf16* H2bf   = (__bf16*)(ws + 55574528);    // 1024 x 1024
  __bf16* Wboxbf = (__bf16*)(ws + 57671680);    // 1024 x 384 (324 padded)
  __bf16* Wclsbf = (__bf16*)(ws + 58458112);    // 1024 x 128 (81 padded)

  float* out_box = (float*)d_out;                       // [1024, 324]
  float* out_cls = out_box + (size_t)N_ROIS * N_BOX;    // [1024, 81]

  // 1) RoI pool -> bf16 activations
  roi_pool_kernel<<<N_ROIS, 256, 0, stream>>>(feats, props, Xbf);

  // 2) weight conversions (f32 -> bf16, pad head weights to 64-col multiples)
  conv_pad_kernel<<<(12544 * 1024 + 255) / 256, 256, 0, stream>>>(W1, W1bf, 12544, 1024, 1024);
  conv_pad_kernel<<<(1024 * 1024 + 255) / 256, 256, 0, stream>>>(W2, W2bf, 1024, 1024, 1024);
  conv_pad_kernel<<<(1024 * 384 + 255) / 256, 256, 0, stream>>>(Wbox, Wboxbf, 1024, N_BOX, 384);
  conv_pad_kernel<<<(1024 * 128 + 255) / 256, 256, 0, stream>>>(Wcls, Wclsbf, 1024, N_CLS, 128);

  // 3) FC1 (relu, bf16 out), FC2 (relu, bf16 out): 128x64 tiles -> 16x8 = 128 WGs
  gemm_bf16_wmma<true,  true ><<<dim3(16, 8), 256, 0, stream>>>(Xbf,  W1bf, b1, H1bf, 12544, 1024, 1024);
  gemm_bf16_wmma<true,  true ><<<dim3(16, 8), 256, 0, stream>>>(H1bf, W2bf, b2, H2bf, 1024, 1024, 1024);

  // 4) heads (f32 out, packed leading dims 324 / 81)
  gemm_bf16_wmma<false, false><<<dim3(6, 8), 256, 0, stream>>>(H2bf, Wboxbf, bbox, out_box, 1024, 384, N_BOX);
  gemm_bf16_wmma<false, false><<<dim3(2, 8), 256, 0, stream>>>(H2bf, Wclsbf, bcls, out_cls, 1024, 128, N_CLS);
}
